// ECHOEncoder_41790031790248
// MI455X (gfx1250) — matrix operand order repaired and checked by reading, not compiled, not run on hardware
//
#include <hip/hip_runtime.h>

typedef __attribute__((ext_vector_type(16))) _Float16 v16h;
typedef __attribute__((ext_vector_type(8)))  float    v8f;

// ---------------------------------------------------------------------------
// Weight packing: f32 row-major [K,N] -> f16 WMMA B-fragment order.
// Per 16x32 (NxK) tile, per lane: 16 contiguous halfs.
// B order used here (32x16 KxN, contiguous k, same as A): lanes 0-15 hold K=0..15 for col=lane,
// lanes 16-31 hold K=16..31 for col=lane-16.
// ---------------------------------------------------------------------------
__global__ __launch_bounds__(256) void pack_b_kernel(const float* __restrict__ W,
                                                     _Float16* __restrict__ out,
                                                     int K, int N) {
  int idx = blockIdx.x * 256 + threadIdx.x;
  if (idx >= K * N) return;
  int i    = idx & 15;          // half within lane
  int lane = (idx >> 4) & 31;   // lane within wave
  int tile = idx >> 9;          // (ks, ct) tile
  int nct  = N >> 4;
  int ks = tile / nct, ct = tile % nct;
  int krow = ks * 32 + ((lane >= 16) ? 16 : 0) + i;
  int col  = ct * 16 + (lane & 15);
  out[idx] = (_Float16)W[krow * N + col];
}

// Pack B' = [W_a1_top | W_a1_bot] : [128 x 64] (separable attention pre-act)
__global__ __launch_bounds__(256) void pack_wa1_kernel(const float* __restrict__ Wa1,
                                                       _Float16* __restrict__ out) {
  const int K = 128, N = 64;
  int idx = blockIdx.x * 256 + threadIdx.x;
  if (idx >= K * N) return;
  int i    = idx & 15;
  int lane = (idx >> 4) & 31;
  int tile = idx >> 9;
  int nct  = N >> 4;
  int ks = tile / nct, ct = tile % nct;
  int krow = ks * 32 + ((lane >= 16) ? 16 : 0) + i;
  int col  = ct * 16 + (lane & 15);
  float v = (col < 32) ? Wa1[krow * 32 + col]
                       : Wa1[(128 + krow) * 32 + (col - 32)];
  out[idx] = (_Float16)v;
}

// ---------------------------------------------------------------------------
// Generic tall-skinny GEMM: C[n,N] = epi(A[n,K] @ B[K,N]).
// f32 A converted to f16 on the fly; B pre-packed f16 fragments; f32 acc via
// v_wmma_f32_16x16x32_f16. 256 threads = 8 waves; block covers 128 rows;
// wave w owns rows base+16w..+15 across all N/16 column tiles.
// EPI: 0 = none, 1 = +bias, 2 = relu(+bias)
// ---------------------------------------------------------------------------
template <int K, int N, int EPI>
__global__ __launch_bounds__(256) void gemm_f16wmma(const float* __restrict__ A,
                                                    const _Float16* __restrict__ Bpk,
                                                    const float* __restrict__ bias,
                                                    float* __restrict__ C, int nrows) {
  constexpr int NCT = N / 16;   // column tiles
  constexpr int NKS = K / 32;   // k steps
  const int lane = threadIdx.x & 31;
  const int wave = threadIdx.x >> 5;
  const long baseRow = (long)blockIdx.x * 128 + wave * 16;

  // A-fragment source row for this lane (clamped so tail loads stay in-bounds;
  // garbage results for clamped rows are never stored).
  long lrow = baseRow + (lane & 15);
  if (lrow > nrows - 1) lrow = nrows - 1;
  const float* __restrict__ Arow = A + lrow * (long)K;
  const int kb = (lane >= 16) ? 16 : 0;  // contiguous k split (matches packed B)

  v8f zero = {};
  v8f acc[NCT];
#pragma unroll
  for (int t = 0; t < NCT; ++t) acc[t] = zero;

#pragma unroll
  for (int ks = 0; ks < NKS; ++ks) {
    const int koff = ks * 32 + kb;
    float4 f0 = *(const float4*)(Arow + koff + 0);
    float4 f1 = *(const float4*)(Arow + koff + 4);
    float4 f2 = *(const float4*)(Arow + koff + 8);
    float4 f3 = *(const float4*)(Arow + koff + 12);
    v16h a;
    a[0]  = (_Float16)f0.x; a[1]  = (_Float16)f0.y;
    a[2]  = (_Float16)f0.z; a[3]  = (_Float16)f0.w;
    a[4]  = (_Float16)f1.x; a[5]  = (_Float16)f1.y;
    a[6]  = (_Float16)f1.z; a[7]  = (_Float16)f1.w;
    a[8]  = (_Float16)f2.x; a[9]  = (_Float16)f2.y;
    a[10] = (_Float16)f2.z; a[11] = (_Float16)f2.w;
    a[12] = (_Float16)f3.x; a[13] = (_Float16)f3.y;
    a[14] = (_Float16)f3.z; a[15] = (_Float16)f3.w;

#pragma unroll
    for (int ct = 0; ct < NCT; ++ct) {
      v16h b = *(const v16h*)(Bpk + ((size_t)(ks * NCT + ct) * 32 + lane) * 16);
      acc[ct] = __builtin_amdgcn_wmma_f32_16x16x32_f16(
          /*neg_a=*/false, a, /*neg_b=*/false, b,
          /*c_mod=*/(short)0, acc[ct], /*reuse_a=*/false, /*reuse_b=*/false);
    }
  }

  // C/D layout: lane l, vgpr g -> row (l<16 ? g : g+8), col = l&15.
  const int rbase = (lane >= 16) ? 8 : 0;
#pragma unroll
  for (int ct = 0; ct < NCT; ++ct) {
    const int col = ct * 16 + (lane & 15);
    float bv = 0.0f;
    if (EPI >= 1) bv = bias[col];
#pragma unroll
    for (int g = 0; g < 8; ++g) {
      long orow = baseRow + rbase + g;
      if (orow < nrows) {
        float v = acc[ct][g] + bv;
        if (EPI == 2) v = v > 0.0f ? v : 0.0f;
        C[orow * (long)N + col] = v;
      }
    }
  }
}

// ---------------------------------------------------------------------------
// Edge pass 1: ex[e] = exp(clip(a2 . leaky(P[src]+Q[dst]+b1))) ; sum_exp[src]+=ex
// One wave per edge; lane k handles attention component k (coalesced 128B rows).
// ---------------------------------------------------------------------------
__global__ __launch_bounds__(256) void edge_attn1(const int* __restrict__ src,
                                                  const int* __restrict__ dst,
                                                  const float* __restrict__ PQ,
                                                  const float* __restrict__ b_a1,
                                                  const float* __restrict__ W_a2,
                                                  const float* __restrict__ b_a2,
                                                  float* __restrict__ ex,
                                                  float* __restrict__ sum_exp, int E) {
  const int lane = threadIdx.x & 31;
  const int wid  = (blockIdx.x * blockDim.x + threadIdx.x) >> 5;
  const int nw   = (gridDim.x * blockDim.x) >> 5;
  const float ba  = b_a1[lane];
  const float wa  = W_a2[lane];
  const float ba2 = b_a2[0];
  for (int e = wid; e < E; e += nw) {
    const int s = src[e], d = dst[e];
    float v = PQ[(long)s * 64 + lane] + PQ[(long)d * 64 + 32 + lane] + ba;
    v = v > 0.0f ? v : 0.2f * v;           // leaky_relu(0.2)
    float m = v * wa;
#pragma unroll
    for (int off = 16; off > 0; off >>= 1) m += __shfl_xor(m, off, 32);
    if (lane == 0) {
      float sc = m + ba2;
      sc = fminf(fmaxf(sc, -20.0f), 10.0f);
      float e_ = __expf(sc);
      ex[e] = e_;
      atomicAdd(&sum_exp[s], e_);
    }
  }
}

// ---------------------------------------------------------------------------
// Edge pass 2: attn = ex/(sum_exp[src]+1e-9); agg[dst] += M[src]*attn
// One wave per edge; 4 x 32-lane chunks cover HID=128 (coalesced 512B rows).
// ---------------------------------------------------------------------------
__global__ __launch_bounds__(256) void edge_attn2(const int* __restrict__ src,
                                                  const int* __restrict__ dst,
                                                  const float* __restrict__ ex,
                                                  const float* __restrict__ sum_exp,
                                                  const float* __restrict__ M,
                                                  float* __restrict__ attn_out,
                                                  float* __restrict__ agg, int E) {
  const int lane = threadIdx.x & 31;
  const int wid  = (blockIdx.x * blockDim.x + threadIdx.x) >> 5;
  const int nw   = (gridDim.x * blockDim.x) >> 5;
  for (int e = wid; e < E; e += nw) {
    const int s = src[e], d = dst[e];
    const float a = ex[e] / (sum_exp[s] + 1e-9f);
    if (lane == 0) attn_out[e] = a;
    const float* __restrict__ Mrow = M + (long)s * 128;
    float* __restrict__ arow = agg + (long)d * 128;
#pragma unroll
    for (int j = 0; j < 4; ++j) {
      const int k = lane + 32 * j;
      atomicAdd(&arow[k], Mrow[k] * a);
    }
  }
}

// S = tanh(S + agg); re-zero agg / sum_exp for the next diffusion step.
__global__ __launch_bounds__(256) void node_update(float* __restrict__ S,
                                                   float* __restrict__ agg,
                                                   float* __restrict__ sum_exp, int n) {
  long idx = (long)blockIdx.x * 256 + threadIdx.x;
  if (idx >= (long)n * 128) return;
  float v = S[idx] + agg[idx];
  S[idx] = tanhf(v);
  agg[idx] = 0.0f;
  if ((idx & 127) == 0) sum_exp[idx >> 7] = 0.0f;
}

__global__ __launch_bounds__(256) void zero_f(float* __restrict__ p, long count) {
  long idx = (long)blockIdx.x * 256 + threadIdx.x;
  if (idx < count) p[idx] = 0.0f;
}

// ---------------------------------------------------------------------------
extern "C" void kernel_launch(void* const* d_in, const int* in_sizes, int n_in,
                              void* d_out, int out_size, void* d_ws, size_t ws_size,
                              hipStream_t stream) {
  const float* X      = (const float*)d_in[0];   // [n,256]
  const int*   idx    = (const int*)  d_in[1];   // [2,E]
  const float* W_lin  = (const float*)d_in[2];   // [256,128]
  const float* b_lin  = (const float*)d_in[3];   // [128]
  const float* W_out  = (const float*)d_in[4];   // [128,128]
  const float* b_out  = (const float*)d_in[5];   // [128]
  const float* W_a1   = (const float*)d_in[6];   // [256,32]
  const float* b_a1   = (const float*)d_in[7];   // [32]
  const float* W_a2   = (const float*)d_in[8];   // [32,1]
  const float* b_a2   = (const float*)d_in[9];   // [1]
  const float* W_node = (const float*)d_in[10];  // [128,128]

  const int n = in_sizes[0] / 256;
  const int E = in_sizes[1] / 2;
  const int* src = idx;
  const int* dst = idx + E;

  // ---- workspace bump allocator (256B aligned) ----
  char* ws = (char*)d_ws;
  size_t off = 0;
  auto alloc = [&](size_t bytes) -> void* {
    off = (off + 255) & ~(size_t)255;
    void* p = ws + off;
    off += bytes;
    return p;
  };
  float*    S       = (float*)alloc((size_t)n * 128 * sizeof(float));
  float*    Mbuf    = (float*)alloc((size_t)n * 128 * sizeof(float));
  float*    agg     = (float*)alloc((size_t)n * 128 * sizeof(float));
  float*    PQ      = (float*)alloc((size_t)n * 64  * sizeof(float));
  float*    sum_exp = (float*)alloc((size_t)n       * sizeof(float));
  float*    ex      = (float*)alloc((size_t)E       * sizeof(float));
  _Float16* pkWlin  = (_Float16*)alloc((size_t)256 * 128 * sizeof(_Float16));
  _Float16* pkWa1   = (_Float16*)alloc((size_t)128 * 64  * sizeof(_Float16));
  _Float16* pkWnode = (_Float16*)alloc((size_t)128 * 128 * sizeof(_Float16));
  _Float16* pkWout  = (_Float16*)alloc((size_t)128 * 128 * sizeof(_Float16));

  float* outF = (float*)d_out;            // [n,128]
  float* attnF = outF + (size_t)n * 128;  // [E]

  // ---- pack weights to f16 fragment order ----
  pack_b_kernel<<<(256 * 128 + 255) / 256, 256, 0, stream>>>(W_lin, pkWlin, 256, 128);
  pack_wa1_kernel<<<(128 * 64 + 255) / 256, 256, 0, stream>>>(W_a1, pkWa1);
  pack_b_kernel<<<(128 * 128 + 255) / 256, 256, 0, stream>>>(W_node, pkWnode, 128, 128);
  pack_b_kernel<<<(128 * 128 + 255) / 256, 256, 0, stream>>>(W_out, pkWout, 128, 128);

  const int gblk = (n + 127) / 128;

  // ---- encoder: S = relu(X @ W_lin + b_lin) ----
  gemm_f16wmma<256, 128, 2><<<gblk, 256, 0, stream>>>(X, pkWlin, b_lin, S, n);

  // ---- zero accumulators ----
  zero_f<<<(int)(((long)n * 128 + 255) / 256), 256, 0, stream>>>(agg, (long)n * 128);
  zero_f<<<(n + 255) / 256, 256, 0, stream>>>(sum_exp, n);

  // ---- diffusion steps ----
  for (int step = 0; step < 2; ++step) {
    // PQ = S @ [W_a1_top | W_a1_bot]  (separable attention pre-activation)
    gemm_f16wmma<128, 64, 0><<<gblk, 256, 0, stream>>>(S, pkWa1, nullptr, PQ, n);
    // M = S @ W_node
    gemm_f16wmma<128, 128, 0><<<gblk, 256, 0, stream>>>(S, pkWnode, nullptr, Mbuf, n);
    // edge attention logits + segment sum over src
    edge_attn1<<<4096, 256, 0, stream>>>(src, dst, PQ, b_a1, W_a2, b_a2, ex, sum_exp, E);
    // normalize + scatter messages to dst (attn of the LAST step lands in d_out)
    edge_attn2<<<4096, 256, 0, stream>>>(src, dst, ex, sum_exp, Mbuf, attnF, agg, E);
    // S = tanh(S + agg); reset agg/sum_exp for next step
    node_update<<<(int)(((long)n * 128 + 255) / 256), 256, 0, stream>>>(S, agg, sum_exp, n);
  }

  // ---- out = S @ W_out + b_out ----
  gemm_f16wmma<128, 128, 1><<<gblk, 256, 0, stream>>>(S, pkWout, b_out, outF, n);
}
